// SparseModel_16123307229621
// MI455X (gfx1250) — compile-verified, matrix operand order checked by reading, not run on hardware
//
#include <hip/hip_runtime.h>
#include <hip/hip_bf16.h>

// ---------------------------------------------------------------------------
// MI455X (gfx1250) 3-layer MLP + softmax, bf16 WMMA with double-buffered LDS.
//   h1 = relu(x @ W1^T + b1)      (bf16 in workspace)
//   h2 = relu(h1 @ W2^T + b2)     (bf16 in workspace)
//   out = softmax(h2 @ Wout^T)
// fp32 operands: global->VGPR->cvt->LDS.  bf16 operands: async global->LDS
// (GLOBAL_LOAD_ASYNC_TO_LDS_B128 / ASYNCcnt) when the toolchain provides it.
// ---------------------------------------------------------------------------

#if __has_builtin(__builtin_amdgcn_global_load_async_to_lds_b128)
#define HAVE_ASYNC_LDS 1
#else
#define HAVE_ASYNC_LDS 0
#endif

typedef __attribute__((ext_vector_type(16))) __bf16 v16bf;
typedef __attribute__((ext_vector_type(8)))  float  v8f;
typedef __attribute__((ext_vector_type(4)))  __bf16 v4bf;
typedef __attribute__((ext_vector_type(4)))  int    i4;

// address-space-qualified element types for the async builtin
typedef __attribute__((address_space(1))) i4 gi4;   // global int4
typedef __attribute__((address_space(3))) i4 li4;   // LDS int4

#define BM 128
#define BN 128
#define BK 32
#define BKP 40          // padded LDS row stride (bf16 elems): conflict-free frag reads
#define THREADS 256     // 8 wave32 waves

union Frag { v16bf v; i4 q[2]; };

// C[M x ldc] = A[M x K] * B[Nrows x K]^T (+bias, +relu).
// A is fp32 or bf16 (template); B/bias fp32; C written fp32 or bf16.
template<bool A_BF16, bool OUT_BF16, bool RELU, bool HAS_BIAS>
__global__ __launch_bounds__(THREADS)
void gemm_bt_wmma(const void* __restrict__ Av,
                  const float* __restrict__ B,
                  const float* __restrict__ bias,
                  void* __restrict__ Cv,
                  int K, int Nrows, int Ntrue, int ldc)
{
    __shared__ __bf16 As[2][BM][BKP];   // 2 x 10 KB
    __shared__ __bf16 Bs[2][BN][BKP];   // 2 x 10 KB

    const float*  A32 = (const float*)Av;
    const __bf16* A16 = (const __bf16*)Av;
    float*  C32 = (float*)Cv;
    __bf16* C16 = (__bf16*)Cv;

    constexpr bool USE_ASYNC = A_BF16 && (HAVE_ASYNC_LDS != 0);

    const int tid  = threadIdx.x;
    const int lane = tid & 31;
    const int wid  = tid >> 5;
    const int wm   = (wid >> 2) * 64;
    const int wn   = (wid & 3)  * 32;
    const int mblk = blockIdx.y * BM;
    const int nblk = blockIdx.x * BN;

    // WMMA 16-bit fragment addressing (ISA 7.12.2): lane L holds row L%16;
    // lanes 0-15 take K chunks {0..7,16..23}, lanes 16-31 {8..15,24..31}.
    const int lrow  = lane & 15;
    const int khalf = (lane >> 4) << 3;

    // fp32 staging coords: one float4 per pass, 4 passes covers 128x32.
    const int srow = tid >> 3;            // 0..31
    const int scol = (tid & 7) << 2;      // 0..28 step 4

    const v8f vzero = {0.f,0.f,0.f,0.f,0.f,0.f,0.f,0.f};
    v8f acc[4][2];
#pragma unroll
    for (int m = 0; m < 4; ++m)
#pragma unroll
        for (int n = 0; n < 2; ++n) acc[m][n] = vzero;

    // register staging buffers (loads batched, converted later)
    float4 raf[4];   // A fp32 path
    i4     rah[2];   // A bf16 fallback path (16B chunks)
    float4 rbf[4];   // B fp32

    // ---- A staging: async global->LDS (bf16), no VGPR round-trip ----
    auto stage_A_async = [&](int kt, int buf) {
#if HAVE_ASYNC_LDS
#pragma unroll
        for (int p = 0; p < 2; ++p) {
            const int c   = tid + p * 256;      // 0..511 chunks of 8 bf16
            const int row = c >> 2;
            const int col = (c & 3) << 3;
            const __bf16* gp = &A16[(size_t)(mblk + row) * K + kt * BK + col];
            __bf16*       lp = &As[buf][row][col];
            __builtin_amdgcn_global_load_async_to_lds_b128(
                (gi4*)(void*)gp, (li4*)(void*)lp, 0, 0);
        }
#else
        (void)kt; (void)buf;
#endif
    };
    auto wait_async = [&]() {
#if HAVE_ASYNC_LDS
#if __has_builtin(__builtin_amdgcn_s_wait_asynccnt)
        __builtin_amdgcn_s_wait_asynccnt(0);
#else
        asm volatile("s_wait_asynccnt 0x0" ::: "memory");
#endif
#endif
    };

    // ---- register-path staging (A fp32 / A bf16 fallback / B fp32) ----
    auto load_tiles = [&](int kt) {
        const int k0 = kt * BK;
        if (!USE_ASYNC) {
            if (A_BF16) {
#pragma unroll
                for (int p = 0; p < 2; ++p) {
                    const int c   = tid + p * 256;
                    const int row = c >> 2;
                    const int col = (c & 3) << 3;
                    rah[p] = *(const i4*)&A16[(size_t)(mblk + row) * K + k0 + col];
                }
            } else {
#pragma unroll
                for (int p = 0; p < 4; ++p) {
                    const int row = srow + p * 32;
                    raf[p] = *(const float4*)&A32[(size_t)(mblk + row) * K + k0 + scol];
                }
            }
        }
#pragma unroll
        for (int p = 0; p < 4; ++p) {
            const int row = srow + p * 32;
            const int g   = nblk + row;
            rbf[p] = (g < Nrows) ? *(const float4*)&B[(size_t)g * K + k0 + scol]
                                 : make_float4(0.f, 0.f, 0.f, 0.f);
        }
    };

    auto store_tiles = [&](int buf) {
        if (!USE_ASYNC) {
            if (A_BF16) {
#pragma unroll
                for (int p = 0; p < 2; ++p) {
                    const int c   = tid + p * 256;
                    const int row = c >> 2;
                    const int col = (c & 3) << 3;
                    *(i4*)&As[buf][row][col] = rah[p];
                }
            } else {
#pragma unroll
                for (int p = 0; p < 4; ++p) {
                    const int row = srow + p * 32;
                    v4bf h;
                    h[0] = (__bf16)raf[p].x; h[1] = (__bf16)raf[p].y;
                    h[2] = (__bf16)raf[p].z; h[3] = (__bf16)raf[p].w;
                    *(v4bf*)&As[buf][row][scol] = h;
                }
            }
        }
#pragma unroll
        for (int p = 0; p < 4; ++p) {
            const int row = srow + p * 32;
            v4bf h;
            h[0] = (__bf16)rbf[p].x; h[1] = (__bf16)rbf[p].y;
            h[2] = (__bf16)rbf[p].z; h[3] = (__bf16)rbf[p].w;
            *(v4bf*)&Bs[buf][row][scol] = h;
        }
    };

    const int nk = K / BK;

    // prologue: stage tile 0
    if (USE_ASYNC) stage_A_async(0, 0);
    load_tiles(0);
    store_tiles(0);
    if (USE_ASYNC) wait_async();
    __syncthreads();

    for (int kt = 0; kt < nk; ++kt) {
        const int cur = kt & 1;

        // issue next tile's loads before the math (overlap with WMMA)
        if (kt + 1 < nk) {
            if (USE_ASYNC) stage_A_async(kt + 1, cur ^ 1);
            load_tiles(kt + 1);
        }
        if (kt + 2 < nk) {
            const int kp = (kt + 2) * BK;
            if (A_BF16)
                __builtin_prefetch(&A16[(size_t)(mblk + (tid >> 2)) * K + kp], 0, 1);
            else
                __builtin_prefetch(&A32[(size_t)(mblk + srow) * K + kp + scol], 0, 1);
            const int g = nblk + srow;
            if (g < Nrows)
                __builtin_prefetch(&B[(size_t)g * K + kp + scol], 0, 1);
        }

        // fragments from LDS (current buffer)
        Frag af[4], bfr[2];
#pragma unroll
        for (int m = 0; m < 4; ++m) {
            const int r = wm + m * 16 + lrow;
            af[m].q[0] = *(const i4*)&As[cur][r][khalf];
            af[m].q[1] = *(const i4*)&As[cur][r][khalf + 16];
        }
#pragma unroll
        for (int n = 0; n < 2; ++n) {
            const int r = wn + n * 16 + lrow;
            bfr[n].q[0] = *(const i4*)&Bs[cur][r][khalf];
            bfr[n].q[1] = *(const i4*)&Bs[cur][r][khalf + 16];
        }

        // 8 WMMAs per wave: 64x32 sub-tile, K += 32
#pragma unroll
        for (int m = 0; m < 4; ++m)
#pragma unroll
            for (int n = 0; n < 2; ++n)
                acc[m][n] = __builtin_amdgcn_wmma_f32_16x16x32_bf16(
                    false, af[m].v, false, bfr[n].v,
                    (short)0, acc[m][n], false, false);

        // drain next tile into the other LDS buffer
        if (kt + 1 < nk) {
            store_tiles(cur ^ 1);
            if (USE_ASYNC) wait_async();
        }
        __syncthreads();
    }

    // epilogue: bias + relu, scatter per C/D layout (VGPR r, lane L ->
    // M = r + 8*(L>=16), N = L%16)
#pragma unroll
    for (int m = 0; m < 4; ++m) {
#pragma unroll
        for (int n = 0; n < 2; ++n) {
            const int col   = nblk + wn + n * 16 + (lane & 15);
            const int rbase = mblk + wm + m * 16 + ((lane >> 4) << 3);
            if (col < Ntrue) {
                const float bv = HAS_BIAS ? bias[col] : 0.0f;
#pragma unroll
                for (int r = 0; r < 8; ++r) {
                    float v = acc[m][n][r] + bv;
                    if (RELU) v = fmaxf(v, 0.0f);
                    if (OUT_BF16)
                        C16[(size_t)(rbase + r) * ldc + col] = (__bf16)v;
                    else
                        C32[(size_t)(rbase + r) * ldc + col] = v;
                }
            }
        }
    }
}

// Row softmax over N columns; one block per row.
__global__ __launch_bounds__(256)
void softmax_rows(const float* __restrict__ logits, float* __restrict__ out, int N)
{
    __shared__ float red[256];
    const int row = blockIdx.x;
    const int tid = threadIdx.x;
    const float* in = logits + (size_t)row * N;

    float lmax = -INFINITY;
    for (int c = tid; c < N; c += 256) lmax = fmaxf(lmax, in[c]);
    red[tid] = lmax; __syncthreads();
    for (int s = 128; s > 0; s >>= 1) {
        if (tid < s) red[tid] = fmaxf(red[tid], red[tid + s]);
        __syncthreads();
    }
    const float mx = red[0];
    __syncthreads();

    float lsum = 0.f;
    for (int c = tid; c < N; c += 256) lsum += __expf(in[c] - mx);
    red[tid] = lsum; __syncthreads();
    for (int s = 128; s > 0; s >>= 1) {
        if (tid < s) red[tid] += red[tid + s];
        __syncthreads();
    }
    const float inv = 1.0f / red[0];

    float* o = out + (size_t)row * N;
    for (int c = tid; c < N; c += 256) o[c] = __expf(in[c] - mx) * inv;
}

extern "C" void kernel_launch(void* const* d_in, const int* in_sizes, int n_in,
                              void* d_out, int out_size, void* d_ws, size_t ws_size,
                              hipStream_t stream)
{
    const float* x    = (const float*)d_in[0];   // 1024 x 4096
    const float* W1   = (const float*)d_in[1];   // 4096 x 4096
    const float* b1   = (const float*)d_in[2];   // 4096
    const float* W2   = (const float*)d_in[3];   // 4096 x 4096
    const float* b2   = (const float*)d_in[4];   // 4096
    const float* Wout = (const float*)d_in[5];   // 1000 x 4096

    const int Bsz = 1024, DIN = 4096, DH = 4096, DOUT = 1000;

    __bf16* h1     = (__bf16*)d_ws;                       // 8 MB
    __bf16* h2     = h1 + (size_t)Bsz * DH;               // 8 MB
    float*  logits = (float*)(h2 + (size_t)Bsz * DH);     // 4 MB

    dim3 blk(THREADS);

    // h1 = relu(x @ W1^T + b1), bf16 out
    gemm_bt_wmma<false, true, true, true>
        <<<dim3(DH / BN, Bsz / BM), blk, 0, stream>>>(
            x, W1, b1, h1, DIN, DH, DH, DH);

    // h2 = relu(h1 @ W2^T + b2), bf16 in/out
    gemm_bt_wmma<true, true, true, true>
        <<<dim3(DH / BN, Bsz / BM), blk, 0, stream>>>(
            h1, W2, b2, h2, DH, DH, DH, DH);

    // logits = h2 @ Wout^T, fp32 out (N padded to 1024 in grid, guarded to 1000)
    gemm_bt_wmma<true, false, false, false>
        <<<dim3((DOUT + BN - 1) / BN, Bsz / BM), blk, 0, stream>>>(
            h2, Wout, nullptr, logits, DH, DOUT, DOUT, DOUT);

    // out = softmax(logits) row-wise
    softmax_rows<<<dim3(Bsz), blk, 0, stream>>>(logits, (float*)d_out, DOUT);
}
